// BlocksCore_67053029425661
// MI455X (gfx1250) — matrix operand order repaired and checked by reading, not compiled, not run on hardware
//
#include <hip/hip_runtime.h>
#include <hip/hip_bf16.h>

typedef __attribute__((ext_vector_type(16))) _Float16 v16h;
typedef __attribute__((ext_vector_type(8)))  _Float16 v8h;
typedef __attribute__((ext_vector_type(8)))  float    v8f;

#define TM 64   // block tile rows (4 waves along M)
#define TN 64   // block tile cols (2 waves along N, each wave covers 32)
#define TKQ 32  // K step = WMMA K

// Generic batched GEMM: C[z] = epilogue( A[z](MxK) @ W[z](KxN) )
// f32 in/out, f16 WMMA with fp32 accumulate. Double-buffered LDS software
// pipeline, one workgroup barrier per K-step, 2 wmma per wave per K-step.
// Epilogue: optional per-row scale (scale[row*sRow+z*sZ]) and per-column
// bias (bias[z*biasZ+col]).
__global__ __launch_bounds__(256)
void gemm_f16_wmma(const float* __restrict__ Abase, int lda, long aBatch,
                   const float* __restrict__ Wbase, int ldw, long wBatch,
                   float* __restrict__ Cbase, int ldc, long cBatch,
                   int M, int N, int K,
                   const float* __restrict__ scale, int sRowStride, int sZStride,
                   const float* __restrict__ bias, int biasZStride)
{
    const int z = blockIdx.z;
    const float* A = Abase + (long)z * aBatch;
    const float* W = Wbase + (long)z * wBatch;
    float*       C = Cbase + (long)z * cBatch;

    __shared__ alignas(32) _Float16 As[2][TM][TKQ];  // [buf][m][k]
    __shared__ alignas(32) _Float16 Bs[2][TN][TKQ];  // [buf][n][k] (transposed)

    const int tid  = threadIdx.x;
    const int wave = tid >> 5;
    const int lane = tid & 31;
    const int wm   = wave >> 1;   // 0..3
    const int wn   = wave & 1;    // 0..1

    const int rowBase = blockIdx.y * TM;
    const int colBase = blockIdx.x * TN;

    // cooperative staging coordinates (8 f32 per thread for each of A and B)
    const int ar = tid >> 2;          // 0..63  (A row in tile)
    const int ac = (tid & 3) * 8;     // 0,8,16,24 (A k-offset)
    const int bk = tid >> 3;          // 0..31  (B k-row in tile)
    const int bc = (tid & 7) * 8;     // 0..56  (B n-offset)

    const float* aSrc = A + (long)(rowBase + ar) * lda + ac;
    const float* bSrc = W + (long)bk * ldw + (colBase + bc);

    float aReg[8], bReg[8];

    auto loadRegs = [&](int k0) {
        const float* ap = aSrc + k0;
        #pragma unroll
        for (int i = 0; i < 8; ++i) aReg[i] = ap[i];
        const float* bp = bSrc + (long)k0 * ldw;
        #pragma unroll
        for (int i = 0; i < 8; ++i) bReg[i] = bp[i];
    };
    auto storeLDS = [&](int buf) {
        #pragma unroll
        for (int i = 0; i < 8; ++i) As[buf][ar][ac + i] = (_Float16)aReg[i];
        #pragma unroll
        for (int i = 0; i < 8; ++i) Bs[buf][bc + i][bk] = (_Float16)bReg[i];
    };

    v8f acc0 = {0.f,0.f,0.f,0.f,0.f,0.f,0.f,0.f};
    v8f acc1 = {0.f,0.f,0.f,0.f,0.f,0.f,0.f,0.f};

    loadRegs(0);
    storeLDS(0);
    __syncthreads();

    int buf = 0;
    for (int k0 = 0; k0 < K; k0 += TKQ) {
        const bool hasNext = (k0 + TKQ) < K;
        if (hasNext) loadRegs(k0 + TKQ);   // prefetch next K-slab into registers

        // A fragment (16x32): lane m = lane&15; lanes 0-15 K 0..15,
        // lanes 16-31 K 16..31 -> two 16B LDS reads.
        const int mrow = wm * 16 + (lane & 15);
        const int ka   = (lane >> 4) * 16;
        v8h a0 = *(const v8h*)&As[buf][mrow][ka];
        v8h a1 = *(const v8h*)&As[buf][mrow][ka + 8];
        v16h a;
        #pragma unroll
        for (int j = 0; j < 8; ++j) { a[j] = a0[j]; a[j + 8] = a1[j]; }

        // Two B fragments (32x16 each): lane n = lane&15 within tile;
        // lanes 0-15 K=0..15, lanes 16-31 K=16..31 -> one 32B LDS read each.
        const int ncol = wn * 32 + (lane & 15);
        const int kq   = (lane >> 4) * 16;
        v16h b0 = *(const v16h*)&Bs[buf][ncol][kq];
        v16h b1 = *(const v16h*)&Bs[buf][ncol + 16][kq];

        acc0 = __builtin_amdgcn_wmma_f32_16x16x32_f16(
            false, a, false, b0, (short)0, acc0, false, false);
        acc1 = __builtin_amdgcn_wmma_f32_16x16x32_f16(
            false, a, false, b1, (short)0, acc1, false, false);

        if (hasNext) storeLDS(buf ^ 1);    // write next slab into other buffer
        __syncthreads();                   // single barrier per K-step
        buf ^= 1;
    }

    // C/D layout: VGPR r -> M = (lane>=16 ? 8 : 0) + r, N = lane&15
    const int rbase = rowBase + wm * 16 + ((lane >> 4) << 3);
    const int col0  = colBase + wn * 32 + (lane & 15);
    #pragma unroll
    for (int t = 0; t < 2; ++t) {
        const int col = col0 + t * 16;
        float bcol = bias ? bias[z * biasZStride + col] : 0.0f;
        #pragma unroll
        for (int r = 0; r < 8; ++r) {
            int row = rbase + r;
            float v = (t == 0) ? acc0[r] : acc1[r];
            if (scale) v *= scale[(long)row * sRowStride + z * sZStride];
            C[(long)row * ldc + col] = v + bcol;
        }
    }
}

// Per-row: scores s_k = (q_k . kk1)/8; iatt1 = sigmoid(s), null = 1-iatt1.
// Mask out the 4 blocks with largest null score (ties -> lowest index).
__global__ void score_mask_kernel(const float* __restrict__ q1,   // (B,8,64)
                                  const float* __restrict__ kk1,  // (B,64)
                                  float* __restrict__ scaleA,     // (B,8)
                                  float* __restrict__ maskA,      // (B,8)
                                  int Bn)
{
    int b = blockIdx.x * blockDim.x + threadIdx.x;
    if (b >= Bn) return;
    const float* kp = kk1 + (long)b * 64;
    float nullsc[8];
    #pragma unroll
    for (int k = 0; k < 8; ++k) {
        const float* qp = q1 + (long)b * 512 + k * 64;
        float d = 0.f;
        for (int i = 0; i < 64; ++i) d += qp[i] * kp[i];
        d *= 0.125f;
        float e = __expf(d);            // softmax([0,d]) = [1/(1+e), e/(1+e)]
        scaleA[b * 8 + k] = e / (1.0f + e);
        nullsc[k] = 1.0f / (1.0f + e);
    }
    float m[8];
    bool taken[8];
    #pragma unroll
    for (int k = 0; k < 8; ++k) { m[k] = 1.0f; taken[k] = false; }
    for (int t = 0; t < 4; ++t) {       // top-4 null scores, first-occurrence ties
        int best = 0; float bv = -1e30f;
        for (int k = 0; k < 8; ++k)
            if (!taken[k] && nullsc[k] > bv) { bv = nullsc[k]; best = k; }
        taken[best] = true; m[best] = 0.0f;
    }
    #pragma unroll
    for (int k = 0; k < 8; ++k) maskA[b * 8 + k] = m[k];
}

// GRU gates. gx already = s*(v@wi)+bi, gh = h@wh+bh. i enumerates b*1024+k*128+j.
__global__ void gru_kernel(const float* __restrict__ gx, const float* __restrict__ gh,
                           const float* __restrict__ hx, float* __restrict__ hxnew, int total)
{
    int i = blockIdx.x * blockDim.x + threadIdx.x;
    if (i >= total) return;
    int j  = i & 127;
    int bk = i >> 7;
    const float* gxp = gx + (long)bk * 384;
    const float* ghp = gh + (long)bk * 384;
    float xr = gxp[j], xz = gxp[128 + j], xn = gxp[256 + j];
    float hr = ghp[j], hz = ghp[128 + j], hn = ghp[256 + j];
    float h  = hx[i];
    float r  = 1.f / (1.f + __expf(-(xr + hr)));
    float zz = 1.f / (1.f + __expf(-(xz + hz)));
    float n  = tanhf(xn + r * hn);
    hxnew[i] = (1.f - zz) * n + zz * h;
}

// Tiny 4-head attention over the 8 blocks (head dim 16). One thread per (b,h).
__global__ void attn2_kernel(const float* __restrict__ q2, const float* __restrict__ k2,
                             const float* __restrict__ v2, float* __restrict__ o, int Bn)
{
    int idx = blockIdx.x * blockDim.x + threadIdx.x;
    if (idx >= Bn * 4) return;
    int b = idx >> 2, h = idx & 3;
    const long base = (long)b * 512 + h * 16;
    for (int qb = 0; qb < 8; ++qb) {
        float qv[16];
        #pragma unroll
        for (int d = 0; d < 16; ++d) qv[d] = q2[base + qb * 64 + d];
        float sc[8]; float mx = -1e30f;
        for (int k = 0; k < 8; ++k) {
            float s = 0.f;
            #pragma unroll
            for (int d = 0; d < 16; ++d) s += qv[d] * k2[base + k * 64 + d];
            sc[k] = s * 0.25f;
            mx = fmaxf(mx, sc[k]);
        }
        float sum = 0.f;
        for (int k = 0; k < 8; ++k) { sc[k] = __expf(sc[k] - mx); sum += sc[k]; }
        float inv = 1.f / sum;
        float ov[16];
        #pragma unroll
        for (int d = 0; d < 16; ++d) ov[d] = 0.f;
        for (int k = 0; k < 8; ++k) {
            float p = sc[k] * inv;
            #pragma unroll
            for (int d = 0; d < 16; ++d) ov[d] += p * v2[base + k * 64 + d];
        }
        #pragma unroll
        for (int d = 0; d < 16; ++d) o[base + qb * 64 + d] = ov[d];
    }
}

// att = sigmoid(gate)*tanh(fc); hxnew += att; blend with mask; emit both outputs.
__global__ void final_kernel(const float* __restrict__ gatel, const float* __restrict__ fcl,
                             const float* __restrict__ hxnew, const float* __restrict__ hx,
                             const float* __restrict__ maskA,
                             float* __restrict__ out_hx, float* __restrict__ out_mask, int total)
{
    int i = blockIdx.x * blockDim.x + threadIdx.x;
    if (i >= total) return;
    int bk = i >> 7;
    float g = gatel[i];
    float f = fcl[i];
    float a = (1.f / (1.f + __expf(-g))) * tanhf(f);
    float hn = hxnew[i] + a;
    float m  = maskA[bk];
    out_hx[i]   = m * hn + (1.f - m) * hx[i];
    out_mask[i] = m;
}

extern "C" void kernel_launch(void* const* d_in, const int* in_sizes, int n_in,
                              void* d_out, int out_size, void* d_ws, size_t ws_size,
                              hipStream_t stream)
{
    const float* inp    = (const float*)d_in[0];   // (B,256)
    const float* hx     = (const float*)d_in[1];   // (B,1024)
    const float* Wq1    = (const float*)d_in[3];   // (8,128,64)
    const float* Wk1    = (const float*)d_in[4];   // (2,256,64)
    const float* Wv1    = (const float*)d_in[5];   // (2,256,512)
    const float* Wq2    = (const float*)d_in[6];   // (8,128,64)
    const float* Wk2    = (const float*)d_in[7];
    const float* Wv2    = (const float*)d_in[8];
    const float* fc_w   = (const float*)d_in[9];   // (64,128)
    const float* fc_b   = (const float*)d_in[10];
    const float* gate_w = (const float*)d_in[11];
    const float* gate_b = (const float*)d_in[12];
    const float* gru_wi = (const float*)d_in[13];  // (8,512,384)
    const float* gru_wh = (const float*)d_in[14];  // (8,128,384)
    const float* gru_bi = (const float*)d_in[15];  // (8,384)
    const float* gru_bh = (const float*)d_in[16];

    const int Bn = in_sizes[0] / 256;  // 8192

    float* ws = (float*)d_ws;
    size_t off = 0;
    float* vv1    = ws + off; off += (size_t)Bn * 512;   // x @ Wv1[1]
    float* kk1    = ws + off; off += (size_t)Bn * 64;    // x @ Wk1[1]
    float* q1     = ws + off; off += (size_t)Bn * 512;   // per-block hq @ Wq1
    float* scaleA = ws + off; off += (size_t)Bn * 8;     // iatt1
    float* maskA  = ws + off; off += (size_t)Bn * 8;
    float* gx     = ws + off; off += (size_t)Bn * 3072;  // (B,8,384)
    float* gh     = ws + off; off += (size_t)Bn * 3072;
    float* hxnew  = ws + off; off += (size_t)Bn * 1024;
    // gx region reused after GRU (gatel, fcl take B*1024 each; fcl overlaps gatel and ends at B*3584):
    float* q2     = gx;
    float* k2     = gx + (size_t)Bn * 512;
    float* v2     = gx + (size_t)Bn * 1024;
    float* o      = gx + (size_t)Bn * 1536;
    float* gatel  = gx + (size_t)Bn * 2048;
    float* fcl    = gx + (size_t)Bn * 2560;

    dim3 blk(256);

    // 1) vv1 = x @ Wv1[1]   (M=B, K=256, N=512)
    gemm_f16_wmma<<<dim3(512/TN, Bn/TM, 1), blk, 0, stream>>>(
        inp, 256, 0, Wv1 + 256*512, 512, 0, vv1, 512, 0,
        Bn, 512, 256, nullptr, 0, 0, nullptr, 0);
    // 2) kk1 = x @ Wk1[1]   (N=64)
    gemm_f16_wmma<<<dim3(64/TN, Bn/TM, 1), blk, 0, stream>>>(
        inp, 256, 0, Wk1 + 256*64, 64, 0, kk1, 64, 0,
        Bn, 64, 256, nullptr, 0, 0, nullptr, 0);
    // 3) q1[:,z,:] = hq_z @ Wq1[z]  (8 batches, K=128, N=64)
    gemm_f16_wmma<<<dim3(64/TN, Bn/TM, 8), blk, 0, stream>>>(
        hx, 1024, 128, Wq1, 64, 128*64, q1, 512, 64,
        Bn, 64, 128, nullptr, 0, 0, nullptr, 0);
    // 4) scores -> iatt1 scale + top-k mask
    score_mask_kernel<<<(Bn + 255)/256, blk, 0, stream>>>(q1, kk1, scaleA, maskA, Bn);
    // 5) gx[:,z,:] = iatt1[:,z] * (vv1 @ wi[z]) + bi[z]   (dominant GEMM, K=512, N=384)
    gemm_f16_wmma<<<dim3(384/TN, Bn/TM, 8), blk, 0, stream>>>(
        vv1, 512, 0, gru_wi, 384, (long)512*384, gx, 3072, 384,
        Bn, 384, 512, scaleA, 8, 1, gru_bi, 384);
    // 6) gh[:,z,:] = hq_z @ wh[z] + bh[z]
    gemm_f16_wmma<<<dim3(384/TN, Bn/TM, 8), blk, 0, stream>>>(
        hx, 1024, 128, gru_wh, 384, (long)128*384, gh, 3072, 384,
        Bn, 384, 128, nullptr, 0, 0, gru_bh, 384);
    // 7) GRU elementwise
    gru_kernel<<<(Bn*1024 + 255)/256, blk, 0, stream>>>(gx, gh, hx, hxnew, Bn*1024);
    // 8) q2/k2/v2 projections (8 batches each, K=128, N=64)
    gemm_f16_wmma<<<dim3(64/TN, Bn/TM, 8), blk, 0, stream>>>(
        hxnew, 1024, 128, Wq2, 64, 128*64, q2, 512, 64,
        Bn, 64, 128, nullptr, 0, 0, nullptr, 0);
    gemm_f16_wmma<<<dim3(64/TN, Bn/TM, 8), blk, 0, stream>>>(
        hxnew, 1024, 128, Wk2, 64, 128*64, k2, 512, 64,
        Bn, 64, 128, nullptr, 0, 0, nullptr, 0);
    gemm_f16_wmma<<<dim3(64/TN, Bn/TM, 8), blk, 0, stream>>>(
        hxnew, 1024, 128, Wv2, 64, 128*64, v2, 512, 64,
        Bn, 64, 128, nullptr, 0, 0, nullptr, 0);
    // 9) tiny 4-head attention over the 8 blocks
    attn2_kernel<<<(Bn*4 + 255)/256, blk, 0, stream>>>(q2, k2, v2, o, Bn);
    // 10) gate / fc projections: (B*8,64) @ (64,128) + bias
    gemm_f16_wmma<<<dim3(128/TN, (Bn*8)/TM, 1), blk, 0, stream>>>(
        o, 64, 0, gate_w, 128, 0, gatel, 128, 0,
        Bn*8, 128, 64, nullptr, 0, 0, gate_b, 0);
    gemm_f16_wmma<<<dim3(128/TN, (Bn*8)/TM, 1), blk, 0, stream>>>(
        o, 64, 0, fc_w, 128, 0, fcl, 128, 0,
        Bn*8, 128, 64, nullptr, 0, 0, fc_b, 0);
    // 11) gated add + mask blend; write hx_out and mask_full
    final_kernel<<<(Bn*1024 + 255)/256, blk, 0, stream>>>(
        gatel, fcl, hxnew, hx, maskA,
        (float*)d_out, (float*)d_out + (size_t)Bn*1024, Bn*1024);
}
